// EGNN_27693949125353
// MI455X (gfx1250) — matrix operand order repaired and checked by reading, not compiled, not run on hardware
//
#include <hip/hip_runtime.h>
#include <hip/hip_bf16.h>

#define HD 128
#define EINF 325
#define KT_E1 11   /* ceil(325/32) */
#define KT_128 4
#define KT_256 8
#define NT 8       /* 128 / 16 output column tiles */
#define LAYERS 4

typedef _Float16 v16h __attribute__((ext_vector_type(16)));
typedef _Float16 v8h  __attribute__((ext_vector_type(8)));
typedef float    v8f  __attribute__((ext_vector_type(8)));
union AFrag { v16h v; v8h h[2]; };

__device__ __forceinline__ float silu_f(float u){ return u / (1.0f + __expf(-u)); }

__device__ __forceinline__ void async_copy16_to_lds(unsigned lds_off,
                                                    const void* gaddr) {
  // GLOBAL_LOAD_ASYNC_TO_LDS_B128 (GV mode): LDS[ldsoff] <= MEM[vaddr], 16B/lane,
  // tracked by ASYNCcnt (CDNA5 ISA).
  asm volatile("global_load_async_to_lds_b128 %0, %1, off"
               :: "v"(lds_off), "v"(gaddr) : "memory");
}
__device__ __forceinline__ void wait_asynccnt0() {
  asm volatile("s_wait_asynccnt 0x0" ::: "memory");
}

// ---------------------------------------------------------------------------
// Weight pre-swizzle into WMMA B-fragment order:
// out[kt][nt][lane][i]  (i=0..15)  <-  W[kt*32 + (i>>3)*16 + (lane>>4)*8 + (i&7)][nt*16 + (lane&15)]
// ---------------------------------------------------------------------------
__global__ void k_swizzleB(const float* __restrict__ W, int Kreal,
                           _Float16* __restrict__ out, int total) {
  int idx = blockIdx.x * blockDim.x + threadIdx.x;
  if (idx >= total) return;
  int i    = idx & 15;
  int lane = (idx >> 4) & 31;
  int nt   = (idx >> 9) & 7;
  int kt   = idx >> 12;
  int K  = kt * 32 + (i >> 3) * 16 + (lane >> 4) * 8 + (i & 7);
  int Nc = nt * 16 + (lane & 15);
  float v = (K < Kreal) ? W[K * HD + Nc] : 0.0f;
  out[idx] = (_Float16)v;
}

// ---------------------------------------------------------------------------
// frac_diff (min-image) + degree (count of edges per ei1 target)
// ---------------------------------------------------------------------------
__global__ void k_prep(const float* __restrict__ frac, const int* __restrict__ ei,
                       float* __restrict__ fracd, float* __restrict__ deg, int E) {
  int e = blockIdx.x * blockDim.x + threadIdx.x;
  if (e >= E) return;
  int a = ei[e], b = ei[E + e];
#pragma unroll
  for (int c = 0; c < 3; ++c) {
    float s = frac[b * 3 + c] - frac[a * 3 + c];
    fracd[e * 3 + c] = s - floorf(s + 0.5f + 1e-4f);
  }
  atomicAdd(&deg[b], 1.0f);
}

// lat_ip[g,i,k] = sum_j lat[g,i,j]*lat[g,k,j]
__global__ void k_latip(const float* __restrict__ lat, float* __restrict__ latip, int G9) {
  int idx = blockIdx.x * blockDim.x + threadIdx.x;
  if (idx >= G9) return;
  int g = idx / 9, r = idx - g * 9;
  int i = r / 3, k = r - i * 3;
  float s = 0.f;
#pragma unroll
  for (int j = 0; j < 3; ++j) s += lat[g * 9 + i * 3 + j] * lat[g * 9 + k * 3 + j];
  latip[idx] = s;
}

// ---------------------------------------------------------------------------
// Initial node features: h = [embed[at-1] | sin/cos time emb] @ lat_w + lat_b
// ---------------------------------------------------------------------------
__global__ __launch_bounds__(128) void k_init_h(
    const int* __restrict__ atype, const float* __restrict__ t,
    const float* __restrict__ embed, const float* __restrict__ latw,
    const float* __restrict__ latb, float* __restrict__ h,
    _Float16* __restrict__ hh, int Nn) {
  __shared__ float fin[384];
  int n = blockIdx.x;
  if (n >= Nn) return;
  int c  = threadIdx.x;
  float tv = t[n];
  int at = atype[n] - 1;
  for (int j = c; j < 384; j += 128) {
    float v;
    if (j < HD) {
      v = embed[at * HD + j];
    } else {
      int k  = j - HD;
      int kk = (k < 128) ? k : (k - 128);
      float fr = __expf(-(float)kk * (9.210340371976184f / 127.0f));
      float te = tv * fr;
      v = (k < 128) ? __sinf(te) : __cosf(te);
    }
    fin[j] = v;
  }
  __syncthreads();
  float acc = latb[c];
#pragma unroll 4
  for (int j = 0; j < 384; ++j) acc += fin[j] * latw[j * HD + c];
  h[n * HD + c]  = acc;
  hh[n * HD + c] = (_Float16)acc;
}

// ---------------------------------------------------------------------------
// Edge MLP (WMMA):  block = 128 threads = 4 waves, 16 edges per wave.
// h-row gathers staged via async global->LDS (ASYNCcnt); dis/lat_ip via VALU.
// GEMM1: [16x352]x[352x128] -> silu -> LDS; GEMM2: [16x128]x[128x128] -> silu.
// gateMode==0: atomicAdd ef into aggsum[ei1]
// gateMode==1: GEMM3 ef x coord_w1 -> silu -> dot coord_w2 -> scatter gate.
// ---------------------------------------------------------------------------
__global__ __launch_bounds__(128) void k_edge(
    const _Float16* __restrict__ hh, const int* __restrict__ ei,
    const int* __restrict__ n2g, const float* __restrict__ latip,
    const float* __restrict__ fracd,
    const _Float16* __restrict__ sw1, const _Float16* __restrict__ sw2,
    const float* __restrict__ b1, const float* __restrict__ b2,
    const _Float16* __restrict__ swc1, const float* __restrict__ cb1,
    const float* __restrict__ cw2,
    float* __restrict__ aggsum, float* __restrict__ outsum,
    int E, int gateMode) {
  __shared__ __align__(16) _Float16 As[4][16][352];
  __shared__ __align__(16) _Float16 mid[4][16][HD];

  const int tid  = threadIdx.x;
  const int lane = tid & 31, wid = tid >> 5;
  const int lo   = lane & 15, hi = lane >> 4;
  const int base = blockIdx.x * 64;

  // ---- stage h[ei0] | h[ei1] (k in [0,256)) via async 16B global->LDS ----
  {
    const unsigned ldsbase = (unsigned)(unsigned long long)(&As[0][0][0]);
    for (int ci = tid; ci < 64 * 32; ci += 128) {   // 32 8-f16 chunks per edge
      int el = ci >> 5;
      int k  = (ci & 31) * 8;
      int e  = base + el;
      unsigned dst = ldsbase + (unsigned)(el * 352 + k) * 2u;
      if (e < E) {
        int node = (k < HD) ? ei[e] : ei[E + e];
        async_copy16_to_lds(dst, hh + node * HD + (k & (HD - 1)));
      } else {
        v8h z;
#pragma unroll
        for (int q = 0; q < 8; ++q) z[q] = (_Float16)0.0f;
        *(v8h*)(&As[0][0][0] + el * 352 + k) = z;
      }
    }
  }
  // ---- stage lat_ip + dis (k in [256,352)) via VALU (overlaps async) ----
  for (int idx = tid; idx < 64 * 96; idx += 128) {
    int el = idx / 96, kk = idx - el * 96;
    int k  = 256 + kk;
    int e  = base + el;
    _Float16 v = (_Float16)0.0f;
    if (e < E) {
      if (k < 265) {
        v = (_Float16)latip[n2g[ei[e]] * 9 + (k - 256)];
      } else if (k < EINF) {
        int kp = k - 265;
        int j  = (kp < 30) ? kp : kp - 30;
        int c  = j / 10;
        float de = fracd[e * 3 + c] * (6.283185307179586f * (float)(j - c * 10));
        float s  = (kp < 30) ? __sinf(de) : __cosf(de);
        v = (_Float16)s;
      }
    }
    As[el >> 4][el & 15][k] = v;
  }
  wait_asynccnt0();
  __syncthreads();

  v8f acc[NT];
#pragma unroll
  for (int nt = 0; nt < NT; ++nt)
#pragma unroll
    for (int q = 0; q < 8; ++q) acc[nt][q] = 0.f;

  // ---- GEMM1: K = 352 ----
  for (int kt = 0; kt < KT_E1; ++kt) {
    AFrag a;
    const _Float16* ap = &As[wid][lo][kt * 32 + hi * 8];
    a.h[0] = *(const v8h*)ap;
    a.h[1] = *(const v8h*)(ap + 16);
    const _Float16* bp = sw1 + ((kt * NT) * 32 + lane) * 16;
#pragma unroll
    for (int nt = 0; nt < NT; ++nt) {
      v16h b = *(const v16h*)(bp + nt * 512);
      acc[nt] = __builtin_amdgcn_wmma_f32_16x16x32_f16(
          false, a.v, false, b, (short)0, acc[nt], false, false);
    }
  }

  // bias + silu -> mid (f16)
  {
    float b1v[NT];
#pragma unroll
    for (int nt = 0; nt < NT; ++nt) b1v[nt] = b1[nt * 16 + lo];
#pragma unroll
    for (int nt = 0; nt < NT; ++nt)
#pragma unroll
      for (int j = 0; j < 8; ++j)
        mid[wid][j + 8 * hi][nt * 16 + lo] = (_Float16)silu_f(acc[nt][j] + b1v[nt]);
  }
  __syncthreads();

#pragma unroll
  for (int nt = 0; nt < NT; ++nt)
#pragma unroll
    for (int q = 0; q < 8; ++q) acc[nt][q] = 0.f;

  // ---- GEMM2: K = 128 ----
  for (int kt = 0; kt < KT_128; ++kt) {
    AFrag a;
    const _Float16* ap = &mid[wid][lo][kt * 32 + hi * 8];
    a.h[0] = *(const v8h*)ap;
    a.h[1] = *(const v8h*)(ap + 16);
    const _Float16* bp = sw2 + ((kt * NT) * 32 + lane) * 16;
#pragma unroll
    for (int nt = 0; nt < NT; ++nt) {
      v16h b = *(const v16h*)(bp + nt * 512);
      acc[nt] = __builtin_amdgcn_wmma_f32_16x16x32_f16(
          false, a.v, false, b, (short)0, acc[nt], false, false);
    }
  }

  float b2v[NT];
#pragma unroll
  for (int nt = 0; nt < NT; ++nt) b2v[nt] = b2[nt * 16 + lo];

  if (!gateMode) {
    // ---- scatter ef into aggsum[ei1] ----
#pragma unroll
    for (int j = 0; j < 8; ++j) {
      int M = j + 8 * hi;
      int e = base + wid * 16 + M;
      if (e < E) {
        int dst = ei[E + e];
#pragma unroll
        for (int nt = 0; nt < NT; ++nt) {
          float efv = silu_f(acc[nt][j] + b2v[nt]);
          atomicAdd(&aggsum[dst * HD + nt * 16 + lo], efv);
        }
      }
    }
  } else {
    // ---- fused coordinate gate: ef -> As (f16) -> GEMM3 -> dot cw2 ----
#pragma unroll
    for (int nt = 0; nt < NT; ++nt)
#pragma unroll
      for (int j = 0; j < 8; ++j)
        As[wid][j + 8 * hi][nt * 16 + lo] =
            (_Float16)silu_f(acc[nt][j] + b2v[nt]);
    __syncthreads();

#pragma unroll
    for (int nt = 0; nt < NT; ++nt)
#pragma unroll
      for (int q = 0; q < 8; ++q) acc[nt][q] = 0.f;

    for (int kt = 0; kt < KT_128; ++kt) {
      AFrag a;
      const _Float16* ap = &As[wid][lo][kt * 32 + hi * 8];
      a.h[0] = *(const v8h*)ap;
      a.h[1] = *(const v8h*)(ap + 16);
      const _Float16* bp = swc1 + ((kt * NT) * 32 + lane) * 16;
#pragma unroll
      for (int nt = 0; nt < NT; ++nt) {
        v16h b = *(const v16h*)(bp + nt * 512);
        acc[nt] = __builtin_amdgcn_wmma_f32_16x16x32_f16(
            false, a.v, false, b, (short)0, acc[nt], false, false);
      }
    }

    float p[8];
#pragma unroll
    for (int j = 0; j < 8; ++j) p[j] = 0.f;
#pragma unroll
    for (int nt = 0; nt < NT; ++nt) {
      int Ncol = nt * 16 + lo;
      float bb = cb1[Ncol];
      float wv = cw2[Ncol];
#pragma unroll
      for (int j = 0; j < 8; ++j) p[j] += silu_f(acc[nt][j] + bb) * wv;
    }
    // cross-lane reduce via LDS (overlay on mid; all mid reads done at barrier)
    float* red = (float*)&mid[0][0][0];  // [4][16][16] floats
#pragma unroll
    for (int j = 0; j < 8; ++j)
      red[(wid * 16 + j + 8 * hi) * 16 + lo] = p[j];
    __syncthreads();
    if (lane < 16) {
      int M = lane;
      const float* r = red + (wid * 16 + M) * 16;
      float g = 0.f;
#pragma unroll
      for (int q = 0; q < 16; ++q) g += r[q];
      int e = base + wid * 16 + M;
      if (e < E) {
        int dst = ei[E + e];
#pragma unroll
        for (int c = 0; c < 3; ++c)
          atomicAdd(&outsum[dst * 3 + c], fracd[e * 3 + c] * g);
      }
    }
  }
}

// ---------------------------------------------------------------------------
// Node MLP (WMMA):  input [h | aggsum/deg]  -> two GEMMs -> residual update h
// ---------------------------------------------------------------------------
__global__ __launch_bounds__(128) void k_node(
    float* __restrict__ h, _Float16* __restrict__ hh,
    const float* __restrict__ aggsum, const float* __restrict__ deg,
    const _Float16* __restrict__ sw1, const _Float16* __restrict__ sw2,
    const float* __restrict__ b1, const float* __restrict__ b2, int Nn) {
  __shared__ __align__(16) _Float16 As[4][16][256];
  __shared__ __align__(16) _Float16 mid[4][16][HD];

  const int tid  = threadIdx.x;
  const int lane = tid & 31, wid = tid >> 5;
  const int lo   = lane & 15, hi = lane >> 4;
  const int base = blockIdx.x * 64;

  // ---- stage h rows (k<128) via async 16B global->LDS ----
  {
    const unsigned ldsbase = (unsigned)(unsigned long long)(&As[0][0][0]);
    for (int ci = tid; ci < 64 * 16; ci += 128) {  // 16 8-f16 chunks per node
      int el = ci >> 4;
      int k  = (ci & 15) * 8;
      int n  = base + el;
      unsigned dst = ldsbase + (unsigned)(el * 256 + k) * 2u;
      if (n < Nn) {
        async_copy16_to_lds(dst, hh + n * HD + k);
      } else {
        v8h z;
#pragma unroll
        for (int q = 0; q < 8; ++q) z[q] = (_Float16)0.0f;
        *(v8h*)(&As[0][0][0] + el * 256 + k) = z;
      }
    }
  }
  // ---- stage agg = aggsum/deg (k in [128,256)) via VALU ----
  for (int idx = tid; idx < 64 * 128; idx += 128) {
    int el = idx >> 7, k = idx & 127;
    int n  = base + el;
    _Float16 v = (_Float16)0.0f;
    if (n < Nn) {
      float d = deg[n];
      d = (d > 1.f) ? d : 1.f;
      v = (_Float16)(aggsum[n * HD + k] / d);
    }
    As[el >> 4][el & 15][HD + k] = v;
  }
  wait_asynccnt0();
  __syncthreads();

  v8f acc[NT];
#pragma unroll
  for (int nt = 0; nt < NT; ++nt)
#pragma unroll
    for (int q = 0; q < 8; ++q) acc[nt][q] = 0.f;

  // GEMM1: K = 256
  for (int kt = 0; kt < KT_256; ++kt) {
    AFrag a;
    const _Float16* ap = &As[wid][lo][kt * 32 + hi * 8];
    a.h[0] = *(const v8h*)ap;
    a.h[1] = *(const v8h*)(ap + 16);
    const _Float16* bp = sw1 + ((kt * NT) * 32 + lane) * 16;
#pragma unroll
    for (int nt = 0; nt < NT; ++nt) {
      v16h b = *(const v16h*)(bp + nt * 512);
      acc[nt] = __builtin_amdgcn_wmma_f32_16x16x32_f16(
          false, a.v, false, b, (short)0, acc[nt], false, false);
    }
  }
  {
    float b1v[NT];
#pragma unroll
    for (int nt = 0; nt < NT; ++nt) b1v[nt] = b1[nt * 16 + lo];
#pragma unroll
    for (int nt = 0; nt < NT; ++nt)
#pragma unroll
      for (int j = 0; j < 8; ++j)
        mid[wid][j + 8 * hi][nt * 16 + lo] = (_Float16)silu_f(acc[nt][j] + b1v[nt]);
  }
  __syncthreads();

#pragma unroll
  for (int nt = 0; nt < NT; ++nt)
#pragma unroll
    for (int q = 0; q < 8; ++q) acc[nt][q] = 0.f;

  // GEMM2: K = 128
  for (int kt = 0; kt < KT_128; ++kt) {
    AFrag a;
    const _Float16* ap = &mid[wid][lo][kt * 32 + hi * 8];
    a.h[0] = *(const v8h*)ap;
    a.h[1] = *(const v8h*)(ap + 16);
    const _Float16* bp = sw2 + ((kt * NT) * 32 + lane) * 16;
#pragma unroll
    for (int nt = 0; nt < NT; ++nt) {
      v16h b = *(const v16h*)(bp + nt * 512);
      acc[nt] = __builtin_amdgcn_wmma_f32_16x16x32_f16(
          false, a.v, false, b, (short)0, acc[nt], false, false);
    }
  }
  float b2v[NT];
#pragma unroll
  for (int nt = 0; nt < NT; ++nt) b2v[nt] = b2[nt * 16 + lo];
#pragma unroll
  for (int j = 0; j < 8; ++j) {
    int M = j + 8 * hi;
    int n = base + wid * 16 + M;
    if (n < Nn) {
#pragma unroll
      for (int nt = 0; nt < NT; ++nt) {
        int Ncol = nt * 16 + lo;
        float hn = h[n * HD + Ncol] + silu_f(acc[nt][j] + b2v[nt]);
        h[n * HD + Ncol]  = hn;
        hh[n * HD + Ncol] = (_Float16)hn;
      }
    }
  }
}

__global__ void k_fin(const float* __restrict__ outsum, const float* __restrict__ deg,
                      float* __restrict__ out, int total) {
  int idx = blockIdx.x * blockDim.x + threadIdx.x;
  if (idx >= total) return;
  int n = idx / 3;
  float d = deg[n];
  d = (d > 1.f) ? d : 1.f;
  out[idx] = outsum[idx] / d;
}

// ---------------------------------------------------------------------------
extern "C" void kernel_launch(void* const* d_in, const int* in_sizes, int n_in,
                              void* d_out, int out_size, void* d_ws, size_t ws_size,
                              hipStream_t stream) {
  const int*   atom_types = (const int*)d_in[0];
  const float* t          = (const float*)d_in[1];
  const float* frac       = (const float*)d_in[2];
  const int*   ei         = (const int*)d_in[3];
  const float* lat        = (const float*)d_in[4];
  const int*   n2g        = (const int*)d_in[5];
  const float* embed      = (const float*)d_in[6];
  const float* lat_w      = (const float*)d_in[7];
  const float* lat_b      = (const float*)d_in[8];
  const float* ew1        = (const float*)d_in[9];
  const float* eb1        = (const float*)d_in[10];
  const float* ew2        = (const float*)d_in[11];
  const float* eb2        = (const float*)d_in[12];
  const float* nw1        = (const float*)d_in[13];
  const float* nb1        = (const float*)d_in[14];
  const float* nw2        = (const float*)d_in[15];
  const float* nb2        = (const float*)d_in[16];
  const float* cw1        = (const float*)d_in[17];
  const float* cb1        = (const float*)d_in[18];
  const float* cw2        = (const float*)d_in[19];

  const int Nn = in_sizes[0];
  const int E  = in_sizes[3] / 2;
  const int G  = in_sizes[4] / 9;

  // --- workspace carve-up (256B aligned) ---
  char* w = (char*)d_ws;
  size_t off = 0;
  auto alloc = [&](size_t bytes) -> void* {
    off = (off + 255) & ~(size_t)255;
    void* p = w + off;
    off += bytes;
    return p;
  };
  float*     fracd   = (float*)alloc((size_t)E * 3 * 4);
  float*     latip   = (float*)alloc((size_t)G * 9 * 4);
  float*     deg     = (float*)alloc((size_t)Nn * 4);
  float*     h       = (float*)alloc((size_t)Nn * HD * 4);
  _Float16*  hh      = (_Float16*)alloc((size_t)Nn * HD * 2);
  float*     aggsum  = (float*)alloc((size_t)Nn * HD * 4);
  float*     outsum  = (float*)alloc((size_t)Nn * 3 * 4);
  const size_t SW_E1   = (size_t)KT_E1  * NT * 32 * 16;   // per-layer halfs
  const size_t SW_K128 = (size_t)KT_128 * NT * 32 * 16;
  const size_t SW_K256 = (size_t)KT_256 * NT * 32 * 16;
  _Float16* sw_ew1 = (_Float16*)alloc(LAYERS * SW_E1 * 2);
  _Float16* sw_ew2 = (_Float16*)alloc(LAYERS * SW_K128 * 2);
  _Float16* sw_nw1 = (_Float16*)alloc(LAYERS * SW_K256 * 2);
  _Float16* sw_nw2 = (_Float16*)alloc(LAYERS * SW_K128 * 2);
  _Float16* sw_cw1 = (_Float16*)alloc(SW_K128 * 2);
  if (off > ws_size) return;  // workspace too small; bail safely

  hipMemsetAsync(deg, 0, (size_t)Nn * 4, stream);
  hipMemsetAsync(outsum, 0, (size_t)Nn * 3 * 4, stream);

  // prep: frac_diff + degree, lattice inner products
  k_prep<<<(E + 255) / 256, 256, 0, stream>>>(frac, ei, fracd, deg, E);
  k_latip<<<(G * 9 + 127) / 128, 128, 0, stream>>>(lat, latip, G * 9);

  // weight swizzles into WMMA B-fragment order
  for (int l = 0; l < LAYERS; ++l) {
    k_swizzleB<<<((int)SW_E1 + 255) / 256, 256, 0, stream>>>(
        ew1 + (size_t)l * EINF * HD, EINF, sw_ew1 + l * SW_E1, (int)SW_E1);
    k_swizzleB<<<((int)SW_K128 + 255) / 256, 256, 0, stream>>>(
        ew2 + (size_t)l * HD * HD, HD, sw_ew2 + l * SW_K128, (int)SW_K128);
    k_swizzleB<<<((int)SW_K256 + 255) / 256, 256, 0, stream>>>(
        nw1 + (size_t)l * 2 * HD * HD, 2 * HD, sw_nw1 + l * SW_K256, (int)SW_K256);
    k_swizzleB<<<((int)SW_K128 + 255) / 256, 256, 0, stream>>>(
        nw2 + (size_t)l * HD * HD, HD, sw_nw2 + l * SW_K128, (int)SW_K128);
  }
  k_swizzleB<<<((int)SW_K128 + 255) / 256, 256, 0, stream>>>(
      cw1, HD, sw_cw1, (int)SW_K128);

  // initial node embedding
  k_init_h<<<Nn, 128, 0, stream>>>(atom_types, t, embed, lat_w, lat_b, h, hh, Nn);

  const int edgeBlocks = (E + 63) / 64;
  const int nodeBlocks = (Nn + 63) / 64;

  // layers 0..2: edge MLP (+aggregate) then node MLP (residual update)
  for (int l = 0; l < LAYERS - 1; ++l) {
    hipMemsetAsync(aggsum, 0, (size_t)Nn * HD * 4, stream);
    k_edge<<<edgeBlocks, 128, 0, stream>>>(
        hh, ei, n2g, latip, fracd,
        sw_ew1 + l * SW_E1, sw_ew2 + l * SW_K128,
        eb1 + l * HD, eb2 + l * HD,
        sw_cw1, cb1, cw2,
        aggsum, outsum, E, /*gateMode=*/0);
    k_node<<<nodeBlocks, 128, 0, stream>>>(
        h, hh, aggsum, deg,
        sw_nw1 + l * SW_K256, sw_nw2 + l * SW_K128,
        nb1 + l * HD, nb2 + l * HD, Nn);
  }
  // layer 3: edge MLP fused with coordinate gate; node update is dead code
  {
    const int l = LAYERS - 1;
    k_edge<<<edgeBlocks, 128, 0, stream>>>(
        hh, ei, n2g, latip, fracd,
        sw_ew1 + l * SW_E1, sw_ew2 + l * SW_K128,
        eb1 + l * HD, eb2 + l * HD,
        sw_cw1, cb1, cw2,
        aggsum, outsum, E, /*gateMode=*/1);
  }
  k_fin<<<(Nn * 3 + 255) / 256, 256, 0, stream>>>(outsum, deg, (float*)d_out, Nn * 3);
}
